// BlockDiagLinear_19490561590005
// MI455X (gfx1250) — compile-verified, matrix operand order checked by reading, not run on hardware
//
#include <hip/hip_runtime.h>

typedef __attribute__((ext_vector_type(16))) __bf16 v16bf;
typedef __attribute__((ext_vector_type(8)))  __bf16 v8bf;
typedef __attribute__((ext_vector_type(4)))  __bf16 v4bf;
typedef __attribute__((ext_vector_type(8)))  float  v8f;

#define N_ROWS 8192
#define D_DIM  4096
#define NBLK   8
#define BLK    512

#define TM 128          // output rows per workgroup
#define TN 128          // output cols per workgroup
#define TK 32           // K per step (one bf16 WMMA depth)
#define LSTRIDE 40      // bf16 elements per LDS row: 32 + 8 pad (keeps 16B align)

union FragAB { v16bf v; v8bf h[2]; };

// ---- CDNA5 async global->LDS copy (ASYNCcnt), 16 bytes per lane ----
__device__ __forceinline__ void async_copy_b128(const void* gsrc, void* ldst) {
    unsigned lds_addr = (unsigned)(size_t)ldst;                 // addr[31:0] = LDS offset
    unsigned long long gaddr = (unsigned long long)(size_t)gsrc;
    asm volatile("global_load_async_to_lds_b128 %0, %1, off"
                 :: "v"(lds_addr), "v"(gaddr) : "memory");
}
__device__ __forceinline__ void wait_asynccnt0() {
    asm volatile("s_wait_asynccnt 0" ::: "memory");
}

// split a float4 into bf16 hi + bf16 lo (residual) quads
__device__ __forceinline__ void split4(const float4 v, v4bf* hi, v4bf* lo) {
    v4bf h = { (__bf16)v.x, (__bf16)v.y, (__bf16)v.z, (__bf16)v.w };
    v4bf l = { (__bf16)(v.x - (float)h[0]), (__bf16)(v.y - (float)h[1]),
               (__bf16)(v.z - (float)h[2]), (__bf16)(v.w - (float)h[3]) };
    *hi = h; *lo = l;
}

// ---- prep: W f32 -> (W_hi, W_lo) bf16, once per launch (8 MB total) ----
__global__ __launch_bounds__(256) void prep_w_kernel(
    const float* __restrict__ w, unsigned short* __restrict__ whi_u,
    unsigned short* __restrict__ wlo_u)
{
    size_t i = ((size_t)blockIdx.x * 256 + threadIdx.x) * 4;
    float4 v = *(const float4*)(w + i);
    v4bf h, l; split4(v, &h, &l);
    *(v4bf*)(whi_u + i) = h;
    *(v4bf*)(wlo_u + i) = l;
}

template <bool PREPPED>
__global__ __launch_bounds__(256) void bdl_main_kernel(
    const float* __restrict__ x,             // [N_ROWS][D_DIM] f32
    const float* __restrict__ w,             // [NBLK][BLK][BLK] f32
    const unsigned short* __restrict__ whi_u,// [NBLK][BLK][BLK] bf16 (if PREPPED)
    const unsigned short* __restrict__ wlo_u,
    const float* __restrict__ bias,          // [D_DIM]
    float* __restrict__ y)                   // [N_ROWS][D_DIM]
{
    __shared__ __align__(16) __bf16 xs_hi[TM * LSTRIDE];
    __shared__ __align__(16) __bf16 xs_lo[TM * LSTRIDE];
    __shared__ __align__(16) __bf16 ws_hi[TN * LSTRIDE];
    __shared__ __align__(16) __bf16 ws_lo[TN * LSTRIDE];

    const int tid   = threadIdx.x;
    const int lane  = tid & 31;
    const int wave  = tid >> 5;          // 0..7
    const int waveM = wave & 1;          // 2 waves along rows
    const int waveN = wave >> 1;         // 4 waves along cols

    const int kblk    = blockIdx.x >> 2;          // diagonal block (0..7)
    const int ct      = blockIdx.x & 3;           // col tile within block
    const int rowbase = blockIdx.y * TM;
    const int wr0     = ct * TN;                  // W row base (= out col in block)
    const int colbase = kblk * BLK + wr0;

    // 16-bit A/B fragment lane mapping: element e -> K = (e>>3)*16 + k0 + (e&7)
    const int frow = lane & 15;
    const int k0   = ((lane >> 4) & 1) * 8;

    v8f acc[4][2];
#pragma unroll
    for (int m = 0; m < 4; ++m)
#pragma unroll
        for (int n = 0; n < 2; ++n)
            acc[m][n] = (v8f){0.f, 0.f, 0.f, 0.f, 0.f, 0.f, 0.f, 0.f};

    const float* xsrc = x + (size_t)rowbase * D_DIM + (size_t)kblk * BLK;
    const float*          wsrc_f = w     + ((size_t)kblk * BLK + wr0) * BLK;
    const unsigned short* whsrc  = whi_u + ((size_t)kblk * BLK + wr0) * BLK;
    const unsigned short* wlsrc  = wlo_u + ((size_t)kblk * BLK + wr0) * BLK;

    for (int kk = 0; kk < BLK; kk += TK) {
        __syncthreads();   // previous iteration's readers done before refill

        if (PREPPED) {
            // ---- W tiles already bf16: async DMA straight into LDS ----
            // 128 rows x 32 K per buffer = 512 chunks of 8 bf16 (16B each)
#pragma unroll
            for (int i = 0; i < 2; ++i) {
                int idx = tid + i * 256;          // 0..511
                int r = idx >> 2, q = idx & 3;
                async_copy_b128(whsrc + (size_t)r * BLK + kk + q * 8,
                                &ws_hi[r * LSTRIDE + q * 8]);
                async_copy_b128(wlsrc + (size_t)r * BLK + kk + q * 8,
                                &ws_lo[r * LSTRIDE + q * 8]);
            }
        } else {
            // ---- fallback: load f32 W tile, split to bf16 hi/lo into LDS ----
#pragma unroll
            for (int i = 0; i < 4; ++i) {
                int idx = tid + i * 256;          // 0..1023
                int r = idx >> 3, q = idx & 7;
                float4 v = *(const float4*)(wsrc_f + (size_t)r * BLK + kk + q * 4);
                v4bf h, l; split4(v, &h, &l);
                *(v4bf*)&ws_hi[r * LSTRIDE + q * 4] = h;
                *(v4bf*)&ws_lo[r * LSTRIDE + q * 4] = l;
            }
        }

        // ---- x tile: load f32, split once per element, store bf16 to LDS
        //      (VALU here overlaps the async W DMA) ----
#pragma unroll
        for (int i = 0; i < 4; ++i) {
            int idx = tid + i * 256;              // 0..1023 float4 chunks
            int r = idx >> 3, q = idx & 7;
            float4 v = *(const float4*)(xsrc + (size_t)r * D_DIM + kk + q * 4);
            v4bf h, l; split4(v, &h, &l);
            *(v4bf*)&xs_hi[r * LSTRIDE + q * 4] = h;
            *(v4bf*)&xs_lo[r * LSTRIDE + q * 4] = l;
        }

        if (PREPPED) wait_asynccnt0();
        __syncthreads();

        // ---- B fragments: pure ds_load_b128, no conversion math ----
        FragAB bh[2], bl[2];
#pragma unroll
        for (int n = 0; n < 2; ++n) {
            const __bf16* p = &ws_hi[(waveN * 32 + n * 16 + frow) * LSTRIDE + k0];
            const __bf16* q = &ws_lo[(waveN * 32 + n * 16 + frow) * LSTRIDE + k0];
            bh[n].h[0] = *(const v8bf*)p;  bh[n].h[1] = *(const v8bf*)(p + 16);
            bl[n].h[0] = *(const v8bf*)q;  bl[n].h[1] = *(const v8bf*)(q + 16);
        }

        // ---- A fragments + 3-term bf16x3 WMMAs ----
#pragma unroll
        for (int m = 0; m < 4; ++m) {
            FragAB ah, al;
            const __bf16* p = &xs_hi[(waveM * 64 + m * 16 + frow) * LSTRIDE + k0];
            const __bf16* q = &xs_lo[(waveM * 64 + m * 16 + frow) * LSTRIDE + k0];
            ah.h[0] = *(const v8bf*)p;  ah.h[1] = *(const v8bf*)(p + 16);
            al.h[0] = *(const v8bf*)q;  al.h[1] = *(const v8bf*)(q + 16);
#pragma unroll
            for (int n = 0; n < 2; ++n) {
                acc[m][n] = __builtin_amdgcn_wmma_f32_16x16x32_bf16(
                    false, ah.v, false, bh[n].v, (short)0, acc[m][n], false, false);
                acc[m][n] = __builtin_amdgcn_wmma_f32_16x16x32_bf16(
                    false, al.v, false, bh[n].v, (short)0, acc[m][n], false, false);
                acc[m][n] = __builtin_amdgcn_wmma_f32_16x16x32_bf16(
                    false, ah.v, false, bl[n].v, (short)0, acc[m][n], false, false);
            }
        }
    }

    // ---- epilogue: bias + store per C/D layout ----
    // lanes 0-15: VGPR j -> M=j ; lanes 16-31: VGPR j -> M=j+8 ; N = lane&15
    const int rsel = (lane >> 4) ? 8 : 0;
#pragma unroll
    for (int n = 0; n < 2; ++n) {
        int col = colbase + waveN * 32 + n * 16 + (lane & 15);
        float bv = bias[col];
#pragma unroll
        for (int m = 0; m < 4; ++m) {
            int row0 = rowbase + waveM * 64 + m * 16 + rsel;
#pragma unroll
            for (int j = 0; j < 8; ++j) {
                y[(size_t)(row0 + j) * D_DIM + col] = acc[m][n][j] + bv;
            }
        }
    }
}

extern "C" void kernel_launch(void* const* d_in, const int* in_sizes, int n_in,
                              void* d_out, int out_size, void* d_ws, size_t ws_size,
                              hipStream_t stream) {
    const float* x    = (const float*)d_in[0];   // [8192, 4096] f32
    const float* wblk = (const float*)d_in[1];   // [8, 512, 512] f32
    const float* bias = (const float*)d_in[2];   // [4096] f32
    float* y = (float*)d_out;                    // [8192, 4096] f32

    const size_t wcount = (size_t)NBLK * BLK * BLK;        // 2,097,152 elements
    const size_t needed = wcount * 2 * sizeof(unsigned short); // hi+lo bf16 = 8 MB
    dim3 grid(NBLK * (BLK / TN), N_ROWS / TM);             // (32, 64)

    if (ws_size >= needed) {
        unsigned short* whi = (unsigned short*)d_ws;
        unsigned short* wlo = whi + wcount;
        prep_w_kernel<<<(unsigned)(wcount / 4 / 256), 256, 0, stream>>>(wblk, whi, wlo);
        bdl_main_kernel<true><<<grid, 256, 0, stream>>>(x, wblk, whi, wlo, bias, y);
    } else {
        bdl_main_kernel<false><<<grid, 256, 0, stream>>>(x, wblk, nullptr, nullptr, bias, y);
    }
}
